// SGAT_8306466751031
// MI455X (gfx1250) — matrix operand order repaired and checked by reading, not compiled, not run on hardware
//
#include <hip/hip_runtime.h>

#define N_NODES 50000
#define N_EDGES 800000
#define D_IN    256
#define D_H     96
#define D_OUT   256
#define D_CAT   448   // 96 + 96 + 256
#define BN_EPS  1e-5f

typedef __attribute__((ext_vector_type(16))) __bf16 v16bf;
typedef __attribute__((ext_vector_type(8)))  float  v8f;

// ---------- helpers ----------
__device__ __forceinline__ unsigned short f2bf_rne(float f) {
    unsigned u = __float_as_uint(f);
    unsigned r = (u + 0x7FFFu + ((u >> 16) & 1u)) >> 16;
    return (unsigned short)r;
}
// order-preserving float <-> uint mapping for atomicMax-based segment_max
__device__ __forceinline__ unsigned f2ord(float f) {
    unsigned u = __float_as_uint(f);
    return (u & 0x80000000u) ? ~u : (u | 0x80000000u);
}
__device__ __forceinline__ float ord2f(unsigned u) {
    unsigned v = (u & 0x80000000u) ? (u & 0x7FFFFFFFu) : ~u;
    return __uint_as_float(v);
}

// ---------- 0: zero-init scratch + output ----------
__global__ void k_init(float* sums512, unsigned* segmax, float* denom, float* out) {
    long i = (long)blockIdx.x * blockDim.x + threadIdx.x;
    long stride = (long)gridDim.x * blockDim.x;
    for (long t = i; t < (long)N_NODES * D_OUT; t += stride) out[t] = 0.f;
    for (long t = i; t < N_NODES; t += stride) { denom[t] = 0.f; segmax[t] = 0u; }
    for (long t = i; t < 512; t += stride) sums512[t] = 0.f;
}

// ---------- 1: column sums / sums-of-squares (coalesced: thread = column) ----------
__global__ void k_colstats(const float* __restrict__ feat, float* __restrict__ sums,
                           float* __restrict__ sumsq) {
    int c = threadIdx.x;                       // 0..255
    float s = 0.f, s2 = 0.f;
    for (int r = blockIdx.x; r < N_NODES; r += gridDim.x) {
        float v = feat[(long)r * D_IN + c];
        s += v; s2 += v * v;
    }
    atomicAdd(&sums[c], s);
    atomicAdd(&sumsq[c], s2);
}

// ---------- 2: finalize mean / rstd ----------
__global__ void k_stats_final(const float* __restrict__ sums, const float* __restrict__ sumsq,
                              float* __restrict__ mean, float* __restrict__ rstd) {
    int c = threadIdx.x;
    float m = sums[c] * (1.f / N_NODES);
    float var = sumsq[c] * (1.f / N_NODES) - m * m;   // biased variance (jnp.var)
    mean[c] = m;
    rstd[c] = rsqrtf(var + BN_EPS);
}

// ---------- 3: normalize + cast to bf16 ----------
__global__ void k_normalize(const float* __restrict__ feat, const float* __restrict__ mean,
                            const float* __restrict__ rstd, const float* __restrict__ gamma,
                            const float* __restrict__ beta, unsigned short* __restrict__ xbf) {
    long i = (long)blockIdx.x * blockDim.x + threadIdx.x;
    long stride = (long)gridDim.x * blockDim.x;
    for (long t = i; t < (long)N_NODES * D_IN; t += stride) {
        int c = (int)(t & (D_IN - 1));
        float x = (feat[t] - mean[c]) * rstd[c] * gamma[c] + beta[c];
        xbf[t] = f2bf_rne(x);
    }
}

// ---------- 4: pack W^T = [Wq|Wk|Wv]^T -> bf16 [448,256] ----------
__global__ void k_pack_w(const float* __restrict__ Wq, const float* __restrict__ Wk,
                         const float* __restrict__ Wv, unsigned short* __restrict__ wt) {
    int c = blockIdx.x;        // 0..447  output column
    int r = threadIdx.x;       // 0..255  input (K) row
    float w;
    if (c < D_H)            w = Wq[(long)r * D_H + c];
    else if (c < 2 * D_H)   w = Wk[(long)r * D_H + (c - D_H)];
    else                    w = Wv[(long)r * D_OUT + (c - 2 * D_H)];
    wt[(long)c * D_IN + r] = f2bf_rne(w);
}

// ---------- 5: WMMA GEMM  qkv[N,448] = x[N,256] @ Wcat[256,448] (+bq on cols<96) ----------
// 4 waves per block; wave -> one 16x16 output tile; 8 x v_wmma_f32_16x16x32_bf16
__global__ void k_gemm_wmma(const unsigned short* __restrict__ xbf,  // [N,256] bf16 bits
                            const unsigned short* __restrict__ wt,   // [448,256] bf16 bits
                            const float* __restrict__ bq,
                            float* __restrict__ qkv) {               // [N,448]
    const int lane = threadIdx.x & 31;
    const int wave = threadIdx.x >> 5;                  // 0..3
    const int ct   = blockIdx.x * 4 + wave;             // col tile 0..27
    const int rt   = blockIdx.y;                        // row tile 0..3124
    const int rowBase = rt * 16;
    const int colBase = ct * 16;

    // A-fragment addressing (16x32 bf16, ISA 7.12.2):
    //   lanes 0-15 : row M=lane,   K = {k0+0..7, k0+16..23}
    //   lanes16-31 : row M=lane-16,K = {k0+8..15, k0+24..31}
    const long aRow  = rowBase + (lane & 15);
    const int  aKoff = (lane >> 4) * 8;
    // B-fragment: lane holds one column of B; K contiguous thanks to W^T layout
    //   lanes 0-15 : col N=lane,    K = {k0+0..7, k0+16..23}
    //   lanes16-31 : col N=lane-16, K = {k0+8..15, k0+24..31}
    const long bCol  = colBase + (lane & 15);
    const int  bKoff = (lane >> 4) * 8;

    v8f acc = {};
#pragma unroll
    for (int k0 = 0; k0 < D_IN; k0 += 32) {
        union { uint4 u[2]; v16bf v; } a, b;
        const uint4* pa = reinterpret_cast<const uint4*>(xbf + aRow * D_IN + k0 + aKoff);
        a.u[0] = pa[0];          // K = k0+aKoff   .. +7
        a.u[1] = pa[2];          // K = k0+aKoff+16.. +23
        const uint4* pb = reinterpret_cast<const uint4*>(wt + bCol * D_IN + k0 + bKoff);
        b.u[0] = pb[0];          // K = k0+bKoff   .. +7
        b.u[1] = pb[2];          // K = k0+bKoff+16.. +23
        acc = __builtin_amdgcn_wmma_f32_16x16x32_bf16(
            /*neg_a=*/false, a.v, /*neg_b=*/false, b.v,
            /*c_mod=*/(short)0, acc, /*reuse_a=*/false, /*reuse_b=*/false);
    }

    // C/D layout: VGPR v, lane L -> M = v + 8*(L>=16), N = L&15
    const int m0  = (lane >> 4) * 8;
    const int col = colBase + (lane & 15);
    const float badd = (col < D_H) ? bq[col] : 0.f;
#pragma unroll
    for (int v = 0; v < 8; ++v) {
        qkv[(long)(rowBase + m0 + v) * D_CAT + col] = acc[v] + badd;
    }
}

// ---------- 6: edge score s = w_attn . sigmoid(q[src]+k[dst]); segment max ----------
__global__ void k_edge_score(const float* __restrict__ qkv, const int* __restrict__ src,
                             const int* __restrict__ dst, const float* __restrict__ w_attn,
                             float* __restrict__ s_arr, unsigned* __restrict__ segmax) {
    int e = blockIdx.x * 8 + (threadIdx.x >> 5);
    if (e >= N_EDGES) return;
    int lane = threadIdx.x & 31;
    const float* q = qkv + (long)src[e] * D_CAT;          // cols 0..95
    const float* k = qkv + (long)dst[e] * D_CAT + D_H;    // cols 96..191
    float acc = 0.f;
#pragma unroll
    for (int h = lane; h < D_H; h += 32) {
        float z = q[h] + k[h];
        float sg = 1.f / (1.f + __expf(-z));
        acc += sg * w_attn[h];
    }
#pragma unroll
    for (int off = 16; off > 0; off >>= 1) acc += __shfl_xor(acc, off, 32);
    if (lane == 0) {
        s_arr[e] = acc;
        atomicMax(&segmax[dst[e]], f2ord(acc));
    }
}

// ---------- 7: ex = exp(s - segmax[dst]); denom[dst] += ex ----------
__global__ void k_edge_exp(const float* __restrict__ s_arr, const int* __restrict__ dst,
                           const unsigned* __restrict__ segmax, float* __restrict__ denom,
                           float* __restrict__ exbuf) {
    int e = blockIdx.x * blockDim.x + threadIdx.x;
    if (e >= N_EDGES) return;
    int d = dst[e];
    float ex = __expf(s_arr[e] - ord2f(segmax[d]));
    exbuf[e] = ex;
    atomicAdd(&denom[d], ex);
}

// ---------- 8: out[dst] += v[src] * (ex/denom[dst])  (thread = column, coalesced) ----------
__global__ void k_scatter(const float* __restrict__ qkv, const int* __restrict__ src,
                          const int* __restrict__ dst, const float* __restrict__ exbuf,
                          const float* __restrict__ denom, float* __restrict__ out) {
    int c = threadIdx.x;                                  // 0..255
    for (int e = blockIdx.x; e < N_EDGES; e += gridDim.x) {
        int d = dst[e];
        float a = exbuf[e] / denom[d];
        float v = qkv[(long)src[e] * D_CAT + 2 * D_H + c];
        atomicAdd(&out[(long)d * D_OUT + c], v * a);
    }
}

extern "C" void kernel_launch(void* const* d_in, const int* in_sizes, int n_in,
                              void* d_out, int out_size, void* d_ws, size_t ws_size,
                              hipStream_t stream) {
    const float* feat   = (const float*)d_in[0];
    const int*   src    = (const int*)  d_in[1];
    const int*   dst    = (const int*)  d_in[2];
    const float* gamma  = (const float*)d_in[3];
    const float* beta   = (const float*)d_in[4];
    const float* Wq     = (const float*)d_in[5];
    const float* bq     = (const float*)d_in[6];
    const float* Wk     = (const float*)d_in[7];
    const float* Wv     = (const float*)d_in[8];
    const float* w_attn = (const float*)d_in[9];
    float* out = (float*)d_out;

    // carve workspace (256B-aligned slices), ~122 MB total
    char* p = (char*)d_ws;
    auto carve = [&](size_t bytes) -> char* {
        char* r = p; p += (bytes + 255) & ~(size_t)255; return r;
    };
    float*          sums   = (float*)carve(512 * sizeof(float));       // sums|sumsq
    float*          sumsq  = sums + 256;
    float*          mean   = (float*)carve(512 * sizeof(float));       // mean|rstd
    float*          rstd   = mean + 256;
    unsigned*       segmax = (unsigned*)carve((size_t)N_NODES * 4);
    float*          denom  = (float*)carve((size_t)N_NODES * 4);
    float*          s_arr  = (float*)carve((size_t)N_EDGES * 4);
    float*          exbuf  = (float*)carve((size_t)N_EDGES * 4);
    float*          qkv    = (float*)carve((size_t)N_NODES * D_CAT * 4);
    unsigned short* xbf    = (unsigned short*)carve((size_t)N_NODES * D_IN * 2);
    unsigned short* wt     = (unsigned short*)carve((size_t)D_CAT * D_IN * 2);

    k_init<<<4096, 256, 0, stream>>>(sums, segmax, denom, out);
    k_colstats<<<512, 256, 0, stream>>>(feat, sums, sumsq);
    k_stats_final<<<1, 256, 0, stream>>>(sums, sumsq, mean, rstd);
    k_normalize<<<4096, 256, 0, stream>>>(feat, mean, rstd, gamma, beta, xbf);
    k_pack_w<<<D_CAT, 256, 0, stream>>>(Wq, Wk, Wv, wt);

    dim3 ggrid(7, N_NODES / 16);          // 28 col tiles / 4 waves, 3125 row tiles
    k_gemm_wmma<<<ggrid, 128, 0, stream>>>(xbf, wt, bq, qkv);

    k_edge_score<<<(N_EDGES + 7) / 8, 256, 0, stream>>>(qkv, src, dst, w_attn, s_arr, segmax);
    k_edge_exp<<<(N_EDGES + 255) / 256, 256, 0, stream>>>(s_arr, dst, segmax, denom, exbuf);
    k_scatter<<<16384, 256, 0, stream>>>(qkv, src, dst, exbuf, denom, out);
}
